// Graphormer_45466523795669
// MI455X (gfx1250) — matrix operand order repaired and checked by reading, not compiled, not run on hardware
//
#include <hip/hip_runtime.h>
#include <hip/hip_bf16.h>
#include <math.h>
#include <stdint.h>

typedef __attribute__((ext_vector_type(16))) _Float16 v16h;
typedef __attribute__((ext_vector_type(8)))  _Float16 v8h;
typedef __attribute__((ext_vector_type(8)))  float    v8f;

#define BB   32
#define NN   512
#define DD   256
#define HH   8
#define DHH  32
#define LL   2
#define ROWS (BB*NN)          /* 16384 */
#define KDIM 256              /* contraction dim of every GEMM here */
#define KC   (KDIM/32)        /* 8 K-chunks */
#define EPSV 1e-5f

static __device__ inline v8f wmma16(v16h a, v16h b, v8f c) {
  // D = A(16x32 f16) * B(32x16 f16) + C(16x16 f32)
  return __builtin_amdgcn_wmma_f32_16x16x32_f16(false, a, false, b, (short)0, c,
                                                false, false);
}
// 16 contiguous halves -> v16h (two b128 loads)
static __device__ inline v16h ld16c(const _Float16* p) {
  return __builtin_shufflevector(*(const v8h*)p, *(const v8h*)(p + 8),
                                 0,1,2,3,4,5,6,7,8,9,10,11,12,13,14,15);
}
// A-fragment: halves [0..7] and [16..23] from p (ISA 16-bit A layout chunks)
static __device__ inline v16h ld16a(const _Float16* p) {
  return __builtin_shufflevector(*(const v8h*)p, *(const v8h*)(p + 16),
                                 0,1,2,3,4,5,6,7,8,9,10,11,12,13,14,15);
}

// ---------------- elementwise converts ----------------
__global__ void k_cvt(const float* __restrict__ in, _Float16* __restrict__ out, int n) {
  int i = blockIdx.x * 256 + threadIdx.x;
  if (i < n) out[i] = (_Float16)in[i];
}

__global__ void k_relu_cvt(const float* __restrict__ in, _Float16* __restrict__ out, int n) {
  int i = blockIdx.x * 256 + threadIdx.x;
  if (i < n) { float v = in[i]; out[i] = (_Float16)(v > 0.f ? v : 0.f); }
}

// ---- pack weight W[Nc,K] (f32) into WMMA-B layout:
// Wp[((nt*KC + kc)*32 + lane)*16 + s] = W[nt*16 + lane%16][kc*32 + k(lane/16, s)] (table k order)
// so each lane's 16-element B fragment is 32 contiguous bytes.
__global__ void k_pack_w(const float* __restrict__ in, _Float16* __restrict__ out, int n) {
  int i = blockIdx.x * 256 + threadIdx.x;        // over Nc*K input elements
  if (i >= n) return;
  int r = i >> 8;                                 // row n (K=256)
  int k = i & 255;
  int nt = r >> 4, j = r & 15, kc = k >> 5, kk = k & 31;
  int lane = (((kk >> 3) & 1) << 4) + j, s = (kk & 7) + ((kk >> 4) << 3);
  out[(((size_t)nt * KC + kc) * 32 + lane) * 16 + s] = (_Float16)in[i];
}

// ---------------- WMMA GEMM: C[M,Nc] = A[M,256] * W[Nc,256]^T + bias ----------------
// No LDS: B fragments come pre-packed from Wp. Each wave computes a 16x64 strip
// (4 n-tiles, A fragment reused 4x). grid = (Nc/64, M/128), block = 256 (8 waves).
__global__ void k_gemm(const _Float16* __restrict__ A, const _Float16* __restrict__ Wp,
                       const float* __restrict__ bias, float* __restrict__ C, int Nc) {
  const int lane = threadIdx.x & 31, wave = threadIdx.x >> 5;
  const int nt0 = blockIdx.x * 4;
  const int mt = blockIdx.y * 8 + wave;
  const int g = lane >> 4, c = lane & 15;
  const int row = mt * 16 + c;
  v8f a0 = {}, a1 = {}, a2 = {}, a3 = {};
  const _Float16* pa0 = A + (size_t)row * KDIM + (g << 3);
  const _Float16* pb0 = Wp + ((size_t)nt0 * KC * 32 + lane) * 16;
#pragma unroll 2
  for (int kc = 0; kc < KC; ++kc) {
    v16h af = ld16a(pa0 + kc * 32);
    const _Float16* pb = pb0 + kc * 512;          // (kc*32 lanes)*16
    v16h b0 = ld16c(pb);
    v16h b1 = ld16c(pb + KC * 512);               // next n-tile
    v16h b2 = ld16c(pb + 2 * KC * 512);
    v16h b3 = ld16c(pb + 3 * KC * 512);
    a0 = wmma16(af, b0, a0);
    a1 = wmma16(af, b1, a1);
    a2 = wmma16(af, b2, a2);
    a3 = wmma16(af, b3, a3);
  }
#pragma unroll
  for (int r = 0; r < 8; ++r) {
    int rr = mt * 16 + r + g * 8;                 // C layout: M=r (lanes 0-15), r+8 (16-31)
    float* cp = C + (size_t)rr * Nc + c;
    cp[(nt0 + 0) * 16] = a0[r] + bias[(nt0 + 0) * 16 + c];
    cp[(nt0 + 1) * 16] = a1[r] + bias[(nt0 + 1) * 16 + c];
    cp[(nt0 + 2) * 16] = a2[r] + bias[(nt0 + 2) * 16 + c];
    cp[(nt0 + 3) * 16] = a3[r] + bias[(nt0 + 3) * 16 + c];
  }
}

// ---------------- BatchNorm (training stats over 16384 rows, 256 cols) ----------------
__global__ void k_bn_partial(const float* __restrict__ Y, float* __restrict__ ps,
                             float* __restrict__ pq) {
  int col = threadIdx.x;
  int chunk = blockIdx.x;                         // 256 chunks of 64 rows
  float s = 0.f, q = 0.f;
  for (int r = 0; r < 64; ++r) {
    float v = Y[(size_t)(chunk * 64 + r) * 256 + col];
    s += v; q += v * v;
  }
  ps[chunk * 256 + col] = s;
  pq[chunk * 256 + col] = q;
}

__global__ void k_bn_final(const float* __restrict__ ps, const float* __restrict__ pq,
                           float* __restrict__ mean, float* __restrict__ rstd) {
  int col = threadIdx.x;
  float s = 0.f, q = 0.f;
  for (int ch = 0; ch < 256; ++ch) { s += ps[ch * 256 + col]; q += pq[ch * 256 + col]; }
  float m = s * (1.0f / (float)ROWS);
  float v = q * (1.0f / (float)ROWS) - m * m;     // biased variance
  mean[col] = m;
  rstd[col] = rsqrtf(v + EPSV);
}

__global__ void k_bn_apply(const float* __restrict__ Y, const float* __restrict__ mean,
                           const float* __restrict__ rstd, const float* __restrict__ g,
                           const float* __restrict__ b, float* __restrict__ out) {
  int i = blockIdx.x * 256 + threadIdx.x;
  int col = i & 255;
  float v = (Y[i] - mean[col]) * rstd[col] * g[col] + b[col];
  out[i] = v >= 0.f ? v : 0.01f * v;              // LeakyReLU(0.01)
}

// ---------------- degree embedding ----------------
__global__ void k_deg(const int* __restrict__ adj, int* __restrict__ deg) {
  int n = blockIdx.x * 256 + threadIdx.x;         // 0..511 (grid.x = 2)
  int b = blockIdx.y;
  int cnt = 0;
  const int* p = adj + (size_t)b * NN * NN + n;
  for (int i = 0; i < NN; ++i) cnt += (p[(size_t)i * NN] != 0) ? 1 : 0;
  deg[b * NN + n] = cnt;
}

__global__ void k_adddeg(float* __restrict__ h, _Float16* __restrict__ h16,
                         const int* __restrict__ deg, const float* __restrict__ demb) {
  int i = blockIdx.x * 256 + threadIdx.x;
  int row = i >> 8, col = i & 255;
  float v = h[i] + demb[(size_t)deg[row] * 256 + col];
  h[i] = v;
  h16[i] = (_Float16)v;
}

// ---------------- QKV split ----------------
// qh: [flat][n][32] row-major (A-layout source for Q, scale folded in)
// kt: [flat][n][32] row-major (K^T B-fragments: lane = key, elems = dh)
// vh: [flat][dh=32][n=512] TRANSPOSED (P*V B-fragments: lane = dh, elems = keys)
__global__ void k_split_qkv(const float* __restrict__ qkv, _Float16* __restrict__ qh,
                            _Float16* __restrict__ kt, _Float16* __restrict__ vh) {
  int i = blockIdx.x * 256 + threadIdx.x;         // ROWS*256 threads
  int bn = i >> 8, hd = i & 255;
  int b = bn >> 9, n = bn & 511;
  int h = hd >> 5, d = hd & 31;
  int flat = b * HH + h;
  size_t rm = ((size_t)flat * NN + n) * DHH + d;  // row-major slot
  const float* src = qkv + (size_t)bn * 768;
  const float scale = 0.17677669529663689f;       // 1/sqrt(32)
  qh[rm] = (_Float16)(src[hd] * scale);
  kt[rm] = (_Float16)src[256 + hd];
  vh[((size_t)flat * DHH + d) * NN + n] = (_Float16)src[512 + hd];
}

// ---------------- flash attention: one wave per (flat=b*H+h, 16-query tile) ----------
__global__ void k_attn(const _Float16* __restrict__ qh, const _Float16* __restrict__ kt,
                       const _Float16* __restrict__ vh, const int* __restrict__ spd,
                       const float* __restrict__ semb, float* __restrict__ out) {
  __shared__ alignas(16) _Float16 Ps[16][32];     // P: C-layout -> A-layout staging
  const int lane = threadIdx.x;
  const int qt = blockIdx.x;                      // 0..31 query tiles
  const int flat = blockIdx.y;                    // 0..255 = b*H + h
  const int b = flat >> 3, h = flat & 7;
  const int icol = flat >> 5, borig = flat & 31;  // faithful concat-over-heads layout
  const int g = lane >> 4, c = lane & 15;

  // Q fragment (A layout), reused across all key chunks
  const _Float16* qp = qh + ((size_t)flat * NN + qt * 16 + c) * DHH + (g << 3);
  v16h qa = ld16a(qp);
  // K^T fragment base: lane%16 = key, elems = dh (table order)
  const _Float16* ktp = kt + ((size_t)flat * NN + c) * DHH + (g << 3);

  float mrow[8], lrow[8];
#pragma unroll
  for (int r = 0; r < 8; ++r) { mrow[r] = -1e30f; lrow[r] = 0.f; }
  v8f o0 = {}, o1 = {};
  const int* spb = spd + (size_t)borig * NN * NN;

  for (int ck = 0; ck < 16; ++ck) {               // 32 keys per chunk
    int k0 = ck * 32;
    // scores: B = K^T, lane%16 = key, elem = dh (table order)
    v16h b0 = ld16a(ktp + (size_t)k0 * DHH);
    v16h b1 = ld16a(ktp + (size_t)(k0 + 16) * DHH);
    v8f z = {};
    v8f s0 = wmma16(qa, b0, z);                   // keys k0..k0+15
    v8f s1 = wmma16(qa, b1, z);                   // keys k0+16..k0+31

    __syncthreads();                              // WAR guard on Ps vs previous chunk
#pragma unroll
    for (int r = 0; r < 8; ++r) {                 // online softmax per query row
      int rq = qt * 16 + r + g * 8;
      const int* sp = spb + (size_t)rq * NN;
      int d0 = sp[k0 + c], d1 = sp[k0 + 16 + c];
      if (ck < 15) __builtin_prefetch(sp + k0 + 32 + c, 0, 0);  // next chunk's bias rows
      float bias0 = (d0 < 0) ? -1.f : semb[d0 * HH + icol];
      float bias1 = (d1 < 0) ? -1.f : semb[d1 * HH + icol];
      float v0 = s0[r] + bias0, v1 = s1[r] + bias1;
      float tm = fmaxf(v0, v1);
      tm = fmaxf(tm, __shfl_xor(tm, 1, 32));
      tm = fmaxf(tm, __shfl_xor(tm, 2, 32));
      tm = fmaxf(tm, __shfl_xor(tm, 4, 32));
      tm = fmaxf(tm, __shfl_xor(tm, 8, 32));      // reduce within 16-lane group
      float mn = fmaxf(mrow[r], tm);
      float corr = __expf(mrow[r] - mn);
      float p0 = __expf(v0 - mn), p1 = __expf(v1 - mn);
      float psum = p0 + p1;
      psum += __shfl_xor(psum, 1, 32);
      psum += __shfl_xor(psum, 2, 32);
      psum += __shfl_xor(psum, 4, 32);
      psum += __shfl_xor(psum, 8, 32);
      lrow[r] = lrow[r] * corr + psum;
      mrow[r] = mn;
      o0[r] *= corr; o1[r] *= corr;
      Ps[r + g * 8][c]      = (_Float16)p0;
      Ps[r + g * 8][16 + c] = (_Float16)p1;
    }
    __syncthreads();

    // P in A layout (16 q x 32 keys); V in B layout straight from global
    v16h pa = ld16a(&Ps[c][g << 3]);
    const _Float16* vp = vh + ((size_t)flat * DHH + c) * NN + k0 + (g << 3);
    v16h vb0 = ld16a(vp);                         // dh 0..15
    v16h vb1 = ld16a(vp + 16 * NN);               // dh 16..31
    o0 = wmma16(pa, vb0, o0);
    o1 = wmma16(pa, vb1, o1);
  }
#pragma unroll
  for (int r = 0; r < 8; ++r) {
    int rq = qt * 16 + r + g * 8;
    float inv = 1.0f / lrow[r];
    float* op = out + ((size_t)b * NN + rq) * DD + h * DHH;
    op[c]      = o0[r] * inv;
    op[16 + c] = o1[r] * inv;
  }
}

// ---------------- residual + LayerNorm (row-wise over 256) ----------------
__global__ void k_addln(float* __restrict__ h, _Float16* __restrict__ h16,
                        const float* __restrict__ y, const float* __restrict__ g,
                        const float* __restrict__ bb) {
  __shared__ float red[16];
  int row = blockIdx.x, tid = threadIdx.x;
  float t = h[(size_t)row * 256 + tid] + y[(size_t)row * 256 + tid];
  float s = t, q = t * t;
#pragma unroll
  for (int m = 1; m < 32; m <<= 1) { s += __shfl_xor(s, m, 32); q += __shfl_xor(q, m, 32); }
  int wave = tid >> 5, lane = tid & 31;
  if (lane == 0) { red[wave] = s; red[8 + wave] = q; }
  __syncthreads();
  if (tid == 0) {
    float S = 0.f, Q = 0.f;
    for (int w = 0; w < 8; ++w) { S += red[w]; Q += red[8 + w]; }
    red[0] = S; red[8] = Q;
  }
  __syncthreads();
  float mean = red[0] * (1.f / 256.f);
  float var  = red[8] * (1.f / 256.f) - mean * mean;
  float rs   = rsqrtf(var + EPSV);
  float o = (t - mean) * rs * g[tid] + bb[tid];
  h[(size_t)row * 256 + tid]   = o;
  h16[(size_t)row * 256 + tid] = (_Float16)o;
}

// ---------------------------------------------------------------------------
extern "C" void kernel_launch(void* const* d_in, const int* in_sizes, int n_in,
                              void* d_out, int out_size, void* d_ws, size_t ws_size,
                              hipStream_t stream) {
  (void)in_sizes; (void)n_in; (void)out_size; (void)ws_size;
  const float* x    = (const float*)d_in[0];
  const int*   adj  = (const int*)d_in[1];
  const int*   spd  = (const int*)d_in[2];
  const float* Wf   = (const float*)d_in[3];
  const float* bf   = (const float*)d_in[4];
  const float* bn1g = (const float*)d_in[5];
  const float* bn1b = (const float*)d_in[6];
  const float* demb = (const float*)d_in[7];
  const float* semb = (const float*)d_in[8];
  const float* Wqkv = (const float*)d_in[9];
  const float* bqkv = (const float*)d_in[10];
  const float* Wo   = (const float*)d_in[11];
  const float* bo   = (const float*)d_in[12];
  const float* ln1g = (const float*)d_in[13];
  const float* ln1b = (const float*)d_in[14];
  const float* W1   = (const float*)d_in[15];
  const float* b1   = (const float*)d_in[16];
  const float* W2   = (const float*)d_in[17];
  const float* b2   = (const float*)d_in[18];
  const float* ln2g = (const float*)d_in[19];
  const float* ln2b = (const float*)d_in[20];
  const float* Win  = (const float*)d_in[21];
  const float* bin  = (const float*)d_in[22];
  const float* bn2g = (const float*)d_in[23];
  const float* bn2b = (const float*)d_in[24];

  char* ws = (char*)d_ws;
  size_t off = 0;
  auto take = [&](size_t bytes) -> void* {
    void* p = ws + off;
    off = (off + bytes + 255) & ~(size_t)255;
    return p;
  };
  _Float16* Wf_p   = (_Float16*)take(65536 * 2);
  _Float16* Wqkv_p = (_Float16*)take(393216 * 2);
  _Float16* Wo_p   = (_Float16*)take(131072 * 2);
  _Float16* W1_p   = (_Float16*)take(131072 * 2);
  _Float16* W2_p   = (_Float16*)take(131072 * 2);
  _Float16* Win_p  = (_Float16*)take(65536 * 2);
  float* mean = (float*)take(256 * 4);
  float* rstd = (float*)take(256 * 4);
  float* psum = (float*)take(256 * 256 * 4);
  float* pqs  = (float*)take(256 * 256 * 4);
  int*   deg  = (int*)take((size_t)ROWS * 4);
  float* hf32 = (float*)take((size_t)ROWS * 256 * 4);
  _Float16* hf16 = (_Float16*)take((size_t)ROWS * 256 * 2);
  float* t0   = (float*)take((size_t)ROWS * 768 * 4);      // qkv f32 / GEMM outputs
  float* t1   = (float*)take((size_t)ROWS * 256 * 4);      // attention output f32
  _Float16* qh = (_Float16*)take((size_t)ROWS * 256 * 2);  // reused for attn-out f16
  _Float16* kt = (_Float16*)take((size_t)ROWS * 256 * 2);  // K transposed; reused for relu f16
  _Float16* vh = (_Float16*)take((size_t)ROWS * 256 * 2);

  const int NE = ROWS * 256;                      // 4,194,304
  dim3 blk256(256), blk32(32);
  dim3 gEW((NE + 255) / 256);

  // weight packing (f32 -> f16 WMMA-B layout) + input convert
  k_pack_w<<<dim3(65536 / 256), blk256, 0, stream>>>(Wf, Wf_p, 65536);
  k_pack_w<<<dim3(393216 / 256), blk256, 0, stream>>>(Wqkv, Wqkv_p, 393216);
  k_pack_w<<<dim3(131072 / 256), blk256, 0, stream>>>(Wo, Wo_p, 131072);
  k_pack_w<<<dim3(131072 / 256), blk256, 0, stream>>>(W1, W1_p, 131072);
  k_pack_w<<<dim3(131072 / 256), blk256, 0, stream>>>(W2, W2_p, 131072);
  k_pack_w<<<dim3(65536 / 256), blk256, 0, stream>>>(Win, Win_p, 65536);
  k_cvt<<<gEW, blk256, 0, stream>>>(x, hf16, NE);

  // lin_first + BN + LeakyReLU
  k_gemm<<<dim3(DD / 64, ROWS / 128), blk256, 0, stream>>>(hf16, Wf_p, bf, t0, DD);
  k_bn_partial<<<dim3(256), blk256, 0, stream>>>(t0, psum, pqs);
  k_bn_final<<<dim3(1), blk256, 0, stream>>>(psum, pqs, mean, rstd);
  k_bn_apply<<<gEW, blk256, 0, stream>>>(t0, mean, rstd, bn1g, bn1b, hf32);

  // degree embedding
  k_deg<<<dim3(2, BB), blk256, 0, stream>>>(adj, deg);
  k_adddeg<<<gEW, blk256, 0, stream>>>(hf32, hf16, deg, demb);

  for (int l = 0; l < LL; ++l) {
    // QKV projection
    k_gemm<<<dim3(768 / 64, ROWS / 128), blk256, 0, stream>>>(
        hf16, Wqkv_p + (size_t)l * 768 * 256, bqkv + l * 768, t0, 768);
    k_split_qkv<<<gEW, blk256, 0, stream>>>(t0, qh, kt, vh);
    // attention with SPD bias (flash, per-wave tile)
    k_attn<<<dim3(NN / 16, BB * HH), blk32, 0, stream>>>(qh, kt, vh, spd, semb, t1);
    // output projection + residual + LN1
    k_cvt<<<gEW, blk256, 0, stream>>>(t1, qh, NE);           // attn-out -> f16 (reuse qh)
    k_gemm<<<dim3(DD / 64, ROWS / 128), blk256, 0, stream>>>(
        qh, Wo_p + (size_t)l * 65536, bo + l * 256, t0, DD);
    k_addln<<<dim3(ROWS), blk256, 0, stream>>>(hf32, hf16, t0, ln1g + l * 256, ln1b + l * 256);
    // FFN + residual + LN2
    k_gemm<<<dim3(DD / 64, ROWS / 128), blk256, 0, stream>>>(
        hf16, W1_p + (size_t)l * 65536, b1 + l * 256, t0, DD);
    k_relu_cvt<<<gEW, blk256, 0, stream>>>(t0, kt, NE);      // relu -> f16 (reuse kt)
    k_gemm<<<dim3(DD / 64, ROWS / 128), blk256, 0, stream>>>(
        kt, W2_p + (size_t)l * 65536, b2 + l * 256, t0, DD);
    k_addln<<<dim3(ROWS), blk256, 0, stream>>>(hf32, hf16, t0, ln2g + l * 256, ln2b + l * 256);
  }

  // lin_in + BN + LeakyReLU -> d_out (float32)
  k_gemm<<<dim3(DD / 64, ROWS / 128), blk256, 0, stream>>>(hf16, Win_p, bin, t0, DD);
  k_bn_partial<<<dim3(256), blk256, 0, stream>>>(t0, psum, pqs);
  k_bn_final<<<dim3(1), blk256, 0, stream>>>(psum, pqs, mean, rstd);
  k_bn_apply<<<gEW, blk256, 0, stream>>>(t0, mean, rstd, bn2g, bn2b, (float*)d_out);
}
